// NonLocalBlock_13709535608883
// MI455X (gfx1250) — matrix operand order repaired and checked by reading, not compiled, not run on hardware
//
#include <hip/hip_runtime.h>
#include <hip/hip_bf16.h>
#include <hip/hip_fp16.h>

typedef __attribute__((ext_vector_type(16))) _Float16 v16h;
typedef __attribute__((ext_vector_type(8)))  _Float16 v8h;
typedef __attribute__((ext_vector_type(8)))  float    v8f;
typedef __attribute__((ext_vector_type(4)))  unsigned int v4u;
typedef __attribute__((ext_vector_type(8)))  int      v8i_;
typedef __attribute__((ext_vector_type(4)))  int      v4i_;

#define B_   4
#define C_   512
#define N_   4096
#define G_   32
#define CG_  16
#define EPS_ 1e-6f

// ---------------------------------------------------------------------------
// WMMA fragment gather helpers (wave32, v_wmma_f32_16x16x32_f16)
//
// A (16x32 f16) layout (ISA 7.12.2): lane L holds row M=L&15.
//   element i: K = (i&7) + (i>=8 ? 16 : 0) + (L>=16 ? 8 : 0)
//   -> two contiguous 16B loads per lane from a row-major [M][K] matrix.
// B (32x16 f16) layout: lane L holds column N=L&15.
//   element i: K = (i&7) + (i>=8 ? 16 : 0) + (L>=16 ? 8 : 0)
//   -> two contiguous 16B loads per lane from an [N][K] (K-contiguous) matrix.
// C/D (16x16 f32): VGPR r: lanes 0-15 -> M=r, lanes 16-31 -> M=r+8; N=L&15.
// ---------------------------------------------------------------------------
__device__ __forceinline__ v16h join16(v8h lo, v8h hi) {
  return __builtin_shufflevector(lo, hi, 0,1,2,3,4,5,6,7,8,9,10,11,12,13,14,15);
}

__device__ __forceinline__ v16h load_afrag(const _Float16* base, int stride, int lane) {
  const _Float16* p = base + (size_t)(lane & 15) * stride + ((lane >> 4) << 3);
  v8h lo = *(const v8h*)p;
  v8h hi = *(const v8h*)(p + 16);
  return join16(lo, hi);
}

__device__ __forceinline__ v16h load_bfrag(const _Float16* base, int stride, int lane) {
  const _Float16* p = base + (size_t)(lane & 15) * stride + ((lane >> 4) << 3);
  v8h lo = *(const v8h*)p;
  v8h hi = *(const v8h*)(p + 16);
  return join16(lo, hi);
}

#define WMMA_F32_F16(A, Bm, Cm) \
  __builtin_amdgcn_wmma_f32_16x16x32_f16(false, (A), false, (Bm), (short)0, (Cm), false, false)

// ---------------------------------------------------------------------------
// Tensor Data Mover: 2-D tile load Global -> LDS (ISA TDM, D# per §8.3/8.4).
// data_size = 2 bytes.  Issued once per wave (EXEC ignored); completion via
// s_wait_tensorcnt (in-order per wave -> counted waits allow double buffering).
//   g0: [1:0]=count=1, [63:32]=lds_addr, [120:64]=global_addr, [127:126]=type=2
//   g1: [17:16]=data_size=1(2B), [79:48]=tensor_dim0, [111:80]=tensor_dim1,
//       [127:112]=tile_dim0, [143:128]=tile_dim1, [207:160]=tensor_dim0_stride
// ---------------------------------------------------------------------------
__device__ __forceinline__ void tdm_load_2d_f16(unsigned lds_byte_addr,
                                                const _Float16* gptr,
                                                unsigned tensor_d0, unsigned tensor_d1,
                                                unsigned tile_d0, unsigned tile_d1,
                                                unsigned stride0_elems) {
  unsigned long long ga = (unsigned long long)(uintptr_t)gptr;
  v4u g0;
  g0[0] = 1u;                                              // count=1, user mode
  g0[1] = lds_byte_addr;                                   // LDS dest (bytes)
  g0[2] = (unsigned)(ga & 0xffffffffu);                    // global_addr lo
  g0[3] = (unsigned)((ga >> 32) & 0x01ffffffu) | (2u << 30); // addr hi | type=2
  v8i_ g1;
  g1[0] = (int)(1u << 16);                                 // data_size=1 -> 2 bytes
  g1[1] = (int)((tensor_d0 & 0xffffu) << 16);              // tensor_dim0[15:0]
  g1[2] = (int)(((tensor_d0 >> 16) & 0xffffu) |
                ((tensor_d1 & 0xffffu) << 16));            // dim0 hi | dim1 lo
  g1[3] = (int)(((tensor_d1 >> 16) & 0xffffu) |
                ((tile_d0 & 0xffffu) << 16));              // dim1 hi | tile_dim0
  g1[4] = (int)(tile_d1 & 0xffffu);                        // tile_dim1 (tile_dim2=0)
  g1[5] = (int)stride0_elems;                              // tensor_dim0_stride lo
  g1[6] = 0;                                               // stride0 hi | stride1 lo
  g1[7] = 0;
  v4i_ gz = {0, 0, 0, 0};
#if __clang_major__ >= 23
  v8i_ gz8 = {0, 0, 0, 0, 0, 0, 0, 0};
  __builtin_amdgcn_tensor_load_to_lds(g0, g1, gz, gz, gz8, 0);
#else
  __builtin_amdgcn_tensor_load_to_lds(g0, g1, gz, gz, 0);
#endif
}

__device__ __forceinline__ unsigned lds_offset_of(const void* p) {
  // generic pointers to LDS carry the byte offset in the low 32 bits
  return (unsigned)(uintptr_t)p;
}

// ---------------------------------------------------------------------------
// Kernel 1: GroupNorm.  One block per (batch, group).  Two-phase: block-reduce
// mean/var over 16 chan x 4096 pos, then normalize and write h_t[B][N][C] f16
// (packed 16B stores; channel-contiguous layout feeds WMMA B-fragments).
// ---------------------------------------------------------------------------
__global__ __launch_bounds__(256) void gn_kernel(const float* __restrict__ x,
                                                 const float* __restrict__ gamma,
                                                 const float* __restrict__ beta,
                                                 _Float16* __restrict__ h_t) {
  __shared__ float s1s[256], s2s[256];
  int bg  = blockIdx.x;
  int b   = bg / G_;
  int g   = bg % G_;
  int tid = threadIdx.x;
  const float* xg = x + ((size_t)b * C_ + g * CG_) * N_;

  float s1 = 0.f, s2 = 0.f;
#pragma unroll
  for (int c = 0; c < CG_; ++c) {
    const float* row = xg + (size_t)c * N_;
    for (int n = tid; n < N_; n += 256) {
      float v = row[n];
      s1 += v;
      s2 += v * v;
    }
  }
  s1s[tid] = s1;
  s2s[tid] = s2;
  __syncthreads();
  for (int off = 128; off > 0; off >>= 1) {
    if (tid < off) { s1s[tid] += s1s[tid + off]; s2s[tid] += s2s[tid + off]; }
    __syncthreads();
  }
  const float inv_cnt = 1.f / (float)(CG_ * N_);
  float mean = s1s[0] * inv_cnt;
  float var  = s2s[0] * inv_cnt - mean * mean;
  float rstd = rsqrtf(var + EPS_);

  float gm[CG_], bt[CG_];
#pragma unroll
  for (int c = 0; c < CG_; ++c) {
    float ga = gamma[g * CG_ + c];
    gm[c] = ga * rstd;
    bt[c] = beta[g * CG_ + c] - mean * rstd * ga;
  }
  for (int n = tid; n < N_; n += 256) {
    v8h lo, hi;
#pragma unroll
    for (int c = 0; c < 8; ++c)
      lo[c] = (_Float16)(xg[(size_t)c * N_ + n] * gm[c] + bt[c]);
#pragma unroll
    for (int c = 0; c < 8; ++c)
      hi[c] = (_Float16)(xg[(size_t)(c + 8) * N_ + n] * gm[c + 8] + bt[c + 8]);
    _Float16* dst = h_t + ((size_t)b * N_ + n) * C_ + g * CG_;
    *(v8h*)dst       = lo;
    *(v8h*)(dst + 8) = hi;
  }
}

// ---------------------------------------------------------------------------
// Kernel 2: convert the four 512x512 f32 weight matrices to f16.
// ---------------------------------------------------------------------------
__global__ __launch_bounds__(256) void wcvt_kernel(const float* __restrict__ wq,
                                                   const float* __restrict__ wk,
                                                   const float* __restrict__ wv,
                                                   const float* __restrict__ wp,
                                                   _Float16* __restrict__ w16) {
  size_t i = (size_t)blockIdx.x * 256 + threadIdx.x;  // 0 .. C*C-1
  const size_t M = (size_t)C_ * C_;
  w16[0 * M + i] = (_Float16)wq[i];
  w16[1 * M + i] = (_Float16)wk[i];
  w16[2 * M + i] = (_Float16)wv[i];
  w16[3 * M + i] = (_Float16)wp[i];
}

// ---------------------------------------------------------------------------
// Kernel 3: 1x1-conv GEMM  out[d,n] = (sum_c W[d,c] * h[c,n] + bias[d]) * scale
// One wave per 16x16 tile, K=512 in 16 WMMA steps.
// transposed=1 -> write [B][N][C] f16 (packed 16B stores, used for q_t / k_t)
// transposed=0 -> write [B][C][N] f16 (used for v)
// ---------------------------------------------------------------------------
__global__ __launch_bounds__(256) void qkv_kernel(const _Float16* __restrict__ h_t,
                                                  const _Float16* __restrict__ w16,
                                                  const float* __restrict__ bias,
                                                  _Float16* __restrict__ outp,
                                                  float scale, int transposed) {
  int lane = threadIdx.x & 31;
  int w    = threadIdx.x >> 5;
  int n0   = (blockIdx.x * 8 + w) * 16;
  int d0   = blockIdx.y * 16;
  int b    = blockIdx.z;
  const _Float16* bbase = h_t + ((size_t)b * N_ + n0) * C_;

  v8f acc = {};
#pragma unroll 4
  for (int k0 = 0; k0 < C_; k0 += 32) {
    v16h a  = load_afrag(w16 + (size_t)d0 * C_ + k0, C_, lane);
    v16h bb = load_bfrag(bbase + k0, C_, lane);
    acc = WMMA_F32_F16(a, bb, acc);
  }

  int lg = lane >> 4;
  int nn = lane & 15;
  if (transposed) {
    v8h o;
#pragma unroll
    for (int r = 0; r < 8; ++r)
      o[r] = (_Float16)((acc[r] + bias[d0 + 8 * lg + r]) * scale);
    *(v8h*)(outp + ((size_t)b * N_ + n0 + nn) * C_ + d0 + 8 * lg) = o;
  } else {
#pragma unroll
    for (int r = 0; r < 8; ++r) {
      int d = d0 + 8 * lg + r;
      outp[((size_t)b * C_ + d) * N_ + n0 + nn] =
          (_Float16)((acc[r] + bias[d]) * scale);
    }
  }
}

// ---------------------------------------------------------------------------
// Kernel 4: flash attention with double-buffered TDM-staged K tiles.
// Block = 8 waves, all on the same batch, covering 4 query tiles x 2 output
// channel halves; the j-loop is in lockstep so the 32x512 f16 K chunk (32KB)
// is DMA'd Global->LDS ONCE per block per chunk (tensor_load_to_lds).
// Ping-pong buffering: the chunk i+1 DMA is issued right after the barrier
// proving buffer i+1's previous contents were consumed, then we wait
// s_wait_tensorcnt<=1 so chunk i is complete while chunk i+1 stays in flight
// behind this iteration's 48 WMMAs (TDM completes in-order per wave).
// Per 32-j chunk and wave:  S: 32 WMMA (K=512), online softmax, P via wave-
// private LDS transpose, O += P x V^T: 16 WMMA (V frags straight from L2).
// q_t is pre-scaled by C^-0.5.
// ---------------------------------------------------------------------------
__global__ __launch_bounds__(256) void attn_kernel(const _Float16* __restrict__ q_t,
                                                   const _Float16* __restrict__ k_t,
                                                   const _Float16* __restrict__ vv,
                                                   _Float16* __restrict__ a_t) {
  __shared__ __align__(16) _Float16 ktile[2][32 * C_];  // ping-pong, 64 KB
  __shared__ __align__(16) _Float16 pbuf[8][16 * 32];   // per-wave P, 8 KB
  int lane = threadIdx.x & 31;
  int w    = threadIdx.x >> 5;
  int bx   = blockIdx.x;                 // 0..255
  int b    = bx >> 6;                    // batch
  int it   = (bx & 63) * 4 + (w >> 1);   // query tile (16 rows)
  int ch   = w & 1;                      // output-channel half
  int i0   = it * 16;
  int c0   = ch * 256;
  int lg   = lane >> 4;
  int nn   = lane & 15;

  const _Float16* qbase = q_t + ((size_t)b * N_ + i0) * C_;
  const _Float16* kbase = k_t + (size_t)b * N_ * C_;
  const _Float16* vbase = vv + (size_t)b * C_ * N_;
  _Float16* lrow = &pbuf[w][0];
  const unsigned kt_lds[2] = { lds_offset_of(&ktile[0][0]),
                               lds_offset_of(&ktile[1][0]) };

  v8f O[16];
#pragma unroll
  for (int t = 0; t < 16; ++t) O[t] = (v8f){};
  float m[8], l[8];
#pragma unroll
  for (int r = 0; r < 8; ++r) { m[r] = -3.0e38f; l[r] = 0.f; }

  // prologue: start DMA of chunk 0 into buffer 0
  if (threadIdx.x < 32) {
    tdm_load_2d_f16(kt_lds[0], kbase,
                    /*tensor_d0=*/C_, /*tensor_d1=*/32,
                    /*tile_d0=*/C_, /*tile_d1=*/32, /*stride0=*/C_);
  }

  for (int j0 = 0; j0 < N_; j0 += 32) {
    int cur = (j0 >> 5) & 1;
    __syncthreads();  // all waves done with buffer cur^1 (chunk j0-32)
    if (threadIdx.x < 32) {
      if (j0 + 32 < N_) {
        // kick chunk j0+32 into the buffer just released, keep it in flight
        tdm_load_2d_f16(kt_lds[cur ^ 1], kbase + (size_t)(j0 + 32) * C_,
                        C_, 32, C_, 32, C_);
        __builtin_amdgcn_s_wait_tensorcnt(1);  // chunk j0 complete (in-order)
      } else {
        __builtin_amdgcn_s_wait_tensorcnt(0);
      }
    }
    __syncthreads();  // staged chunk j0 visible to all waves
    const _Float16* kt = &ktile[cur][0];

    v8f S0 = {}, S1 = {};
#pragma unroll 4
    for (int k0 = 0; k0 < C_; k0 += 32) {
      v16h a  = load_afrag(qbase + k0, C_, lane);
      v16h b0 = load_bfrag(kt + k0, C_, lane);              // rows 0..15
      v16h b1 = load_bfrag(kt + 16 * C_ + k0, C_, lane);    // rows 16..31
      S0 = WMMA_F32_F16(a, b0, S0);
      S1 = WMMA_F32_F16(a, b1, S1);
    }
    // ---- online softmax, per accumulator row (row = r + 8*lg) ----
#pragma unroll
    for (int r = 0; r < 8; ++r) {
      float mx = fmaxf(S0[r], S1[r]);
#pragma unroll
      for (int sh = 8; sh >= 1; sh >>= 1)
        mx = fmaxf(mx, __shfl_xor(mx, sh, 16));
      float mn   = fmaxf(m[r], mx);
      float corr = __expf(m[r] - mn);
      float p0   = __expf(S0[r] - mn);
      float p1   = __expf(S1[r] - mn);
      float rs   = p0 + p1;
#pragma unroll
      for (int sh = 8; sh >= 1; sh >>= 1)
        rs += __shfl_xor(rs, sh, 16);
      l[r] = l[r] * corr + rs;
      m[r] = mn;
#pragma unroll
      for (int t = 0; t < 16; ++t) O[t][r] *= corr;
      // stash P (D-layout: row r+8*lg, cols nn / 16+nn) into wave-private LDS
      lrow[(r + 8 * lg) * 32 + nn]      = (_Float16)p0;
      lrow[(r + 8 * lg) * 32 + 16 + nn] = (_Float16)p1;
    }
    // ---- re-read P as an A-fragment (16x32) ----
    const _Float16* pp = lrow + (size_t)nn * 32 + (lg << 3);
    v8h plo = *(const v8h*)pp;
    v8h phi = *(const v8h*)(pp + 16);
    v16h pA = join16(plo, phi);
    // ---- O += P x V^T for our 256-channel half (V frags from L2) ----
#pragma unroll
    for (int t = 0; t < 16; ++t) {
      v16h bv = load_bfrag(vbase + (size_t)(c0 + 16 * t) * N_ + j0, N_, lane);
      O[t] = WMMA_F32_F16(pA, bv, O[t]);
    }
  }

  // ---- epilogue: normalize by l, write a_t[B][N][C] f16 ----
#pragma unroll
  for (int r = 0; r < 8; ++r) {
    float inv = 1.f / l[r];
    int row   = i0 + r + 8 * lg;
#pragma unroll
    for (int t = 0; t < 16; ++t)
      a_t[((size_t)b * N_ + row) * C_ + c0 + 16 * t + nn] =
          (_Float16)(O[t][r] * inv);
  }
}

// ---------------------------------------------------------------------------
// Kernel 5: projection GEMM + bias + residual:  out = x + Wp @ a + bp
// ---------------------------------------------------------------------------
__global__ __launch_bounds__(256) void proj_kernel(const _Float16* __restrict__ a_t,
                                                   const _Float16* __restrict__ wp16,
                                                   const float* __restrict__ bp,
                                                   const float* __restrict__ x,
                                                   float* __restrict__ out) {
  int lane = threadIdx.x & 31;
  int w    = threadIdx.x >> 5;
  int n0   = (blockIdx.x * 8 + w) * 16;
  int d0   = blockIdx.y * 16;
  int b    = blockIdx.z;
  const _Float16* bbase = a_t + ((size_t)b * N_ + n0) * C_;

  v8f acc = {};
#pragma unroll 4
  for (int k0 = 0; k0 < C_; k0 += 32) {
    v16h a  = load_afrag(wp16 + (size_t)d0 * C_ + k0, C_, lane);
    v16h bb = load_bfrag(bbase + k0, C_, lane);
    acc = WMMA_F32_F16(a, bb, acc);
  }
  int lg = lane >> 4;
  int nn = lane & 15;
#pragma unroll
  for (int r = 0; r < 8; ++r) {
    int d = d0 + 8 * lg + r;
    size_t idx = ((size_t)b * C_ + d) * N_ + n0 + nn;
    out[idx] = x[idx] + bp[d] + acc[r];
  }
}

// ---------------------------------------------------------------------------
extern "C" void kernel_launch(void* const* d_in, const int* in_sizes, int n_in,
                              void* d_out, int out_size, void* d_ws, size_t ws_size,
                              hipStream_t stream) {
  (void)in_sizes; (void)n_in; (void)out_size; (void)ws_size;
  const float* x     = (const float*)d_in[0];
  const float* gamma = (const float*)d_in[1];
  const float* beta  = (const float*)d_in[2];
  const float* wq    = (const float*)d_in[3];
  const float* bq    = (const float*)d_in[4];
  const float* wk    = (const float*)d_in[5];
  const float* bk    = (const float*)d_in[6];
  const float* wv    = (const float*)d_in[7];
  const float* bv    = (const float*)d_in[8];
  const float* wp    = (const float*)d_in[9];
  const float* bp    = (const float*)d_in[10];
  float* out = (float*)d_out;

  const size_t TENS = (size_t)B_ * N_ * C_;   // 8M f16 elements = 16 MB
  const size_t WMAT = (size_t)C_ * C_;        // 256K f16 elements
  char* ws = (char*)d_ws;
  _Float16* h_t = (_Float16*)(ws);
  _Float16* q_t = (_Float16*)(ws + 2 * TENS);
  _Float16* k_t = (_Float16*)(ws + 4 * TENS);
  _Float16* v_m = (_Float16*)(ws + 6 * TENS);
  _Float16* a_t = (_Float16*)(ws + 8 * TENS);
  _Float16* w16 = (_Float16*)(ws + 10 * TENS);

  const float attn_scale = 0.04419417382415922f;  // 512^-0.5

  wcvt_kernel<<<dim3((C_ * C_) / 256), dim3(256), 0, stream>>>(wq, wk, wv, wp, w16);
  gn_kernel<<<dim3(B_ * G_), dim3(256), 0, stream>>>(x, gamma, beta, h_t);

  dim3 ggrid(N_ / 128, C_ / 16, B_);
  qkv_kernel<<<ggrid, dim3(256), 0, stream>>>(h_t, w16 + 0 * WMAT, bq, q_t, attn_scale, 1);
  qkv_kernel<<<ggrid, dim3(256), 0, stream>>>(h_t, w16 + 1 * WMAT, bk, k_t, 1.0f, 1);
  qkv_kernel<<<ggrid, dim3(256), 0, stream>>>(h_t, w16 + 2 * WMAT, bv, v_m, 1.0f, 0);

  attn_kernel<<<dim3(256), dim3(256), 0, stream>>>(q_t, k_t, v_m, a_t);

  proj_kernel<<<ggrid, dim3(256), 0, stream>>>(a_t, w16 + 3 * WMAT, bp, x, out);
}
